// MMAudio_91207925498024
// MI455X (gfx1250) — matrix operand order repaired and checked by reading, compiled, not run on hardware
//
#include <hip/hip_runtime.h>
#include <hip/hip_bf16.h>
#include <math.h>

// ---------------------------------------------------------------------------
// MMAudio-style transformer forward on gfx1250 (MI455X), f16 WMMA + f32 accum.
// B=4 N=1024 VD=1024 D=768 H=12 HD=64 MLP=3072 AD=512 DEPTH=4, T=4096 tokens.
// Global->LDS staging is software-pipelined (next tile held in VGPRs while
// current tile computes from LDS) so global latency hides behind WMMA work.
// ---------------------------------------------------------------------------

typedef __attribute__((ext_vector_type(16))) _Float16 v16h;
typedef __attribute__((ext_vector_type(8)))  _Float16 v8h;
typedef __attribute__((ext_vector_type(8)))  float    v8f;

union U16 { v16h v; v8h h[2]; };

__device__ __forceinline__ float gelu_exact(float x) {
    return 0.5f * x * (1.0f + erff(x * 0.70710678118654752f));
}

// ------------------------------ cast f32 -> f16 ----------------------------
__global__ __launch_bounds__(256) void cast_f16_kernel(const float* __restrict__ src,
                                                       _Float16* __restrict__ dst, int n) {
    int i = blockIdx.x * 256 + threadIdx.x;
    if (i < n) dst[i] = (_Float16)src[i];
}

__global__ __launch_bounds__(256) void cast32to16_kernel(const float* __restrict__ src,
                                                         _Float16* __restrict__ dst, int n) {
    int i = blockIdx.x * 256 + threadIdx.x;
    if (i < n) dst[i] = (_Float16)src[i];
}

// --------------------------- L2 normalize rows -----------------------------
__global__ __launch_bounds__(256) void l2norm_kernel(const float* __restrict__ x,
                                                     _Float16* __restrict__ out, int C) {
    int row = blockIdx.x, t = threadIdx.x;
    const float* xr = x + (size_t)row * C;
    float s = 0.f;
    for (int c = t; c < C; c += 256) { float v = xr[c]; s += v * v; }
    __shared__ float red[256];
    red[t] = s; __syncthreads();
    for (int off = 128; off; off >>= 1) { if (t < off) red[t] += red[t + off]; __syncthreads(); }
    float nrm = sqrtf(red[0]);
    float inv = 1.0f / fmaxf(nrm, 1e-12f);
    for (int c = t; c < C; c += 256) out[(size_t)row * C + c] = (_Float16)(xr[c] * inv);
}

// ------------------------------- LayerNorm ---------------------------------
__global__ __launch_bounds__(256) void ln_kernel(const float* __restrict__ x, int C,
                                                 const float* __restrict__ g,
                                                 const float* __restrict__ b,
                                                 float* __restrict__ out32,
                                                 _Float16* __restrict__ out16,
                                                 int addPE, int gelu) {
    int row = blockIdx.x, t = threadIdx.x;
    const float* xr = x + (size_t)row * C;
    float s = 0.f, s2 = 0.f;
    for (int c = t; c < C; c += 256) { float v = xr[c]; s += v; s2 += v * v; }
    __shared__ float r1[256], r2[256];
    r1[t] = s; r2[t] = s2; __syncthreads();
    for (int off = 128; off; off >>= 1) {
        if (t < off) { r1[t] += r1[t + off]; r2[t] += r2[t + off]; }
        __syncthreads();
    }
    float mu = r1[0] / C;
    float var = r2[0] / C - mu * mu;
    float rs = rsqrtf(var + 1e-5f);
    int n = row & 1023; // token index within sequence (N=1024)
    for (int c = t; c < C; c += 256) {
        float v = (xr[c] - mu) * rs * g[c] + b[c];
        if (gelu) v = gelu_exact(v);
        if (addPE) {
            float i2 = (float)((c >> 1) * 2);
            float freq = __expf(i2 * (-9.210340371976184f / 768.0f)); // -ln(10000)/D
            float ang = (float)n * freq;
            v += (c & 1) ? __cosf(ang) : __sinf(ang);
        }
        size_t o = (size_t)row * C + c;
        if (out32) out32[o] = v;
        if (out16) out16[o] = (_Float16)v;
    }
}

// ------------------------------ WMMA NT GEMM -------------------------------
// C[M,N] = A[M,K](f16,row-major) * W[N,K]^T (f16,row-major) (+bias)(gelu)(+res)
// Block tile 128x64x32, 8 waves, each wave 32x32 via 2x2 v_wmma_f32_16x16x32_f16.
// Next k-tile is prefetched into registers while the current one computes.
#define GBM 128
#define GBN 64
#define GBK 32
#define GBTS (GBN + 8)

__global__ __launch_bounds__(256) void gemm_nt_kernel(
    const _Float16* __restrict__ A, const _Float16* __restrict__ W,
    const float* __restrict__ bias, const float* __restrict__ res,
    float* __restrict__ out32, _Float16* __restrict__ out16,
    int M, int N, int K, int gelu) {
    __shared__ _Float16 As[GBM * GBK];      // 8 KB, row-major rows of 32 halves
    __shared__ _Float16 Bt[GBK * GBTS];     // 4.5 KB, transposed: Bt[k][n]

    int tid = threadIdx.x, lane = tid & 31, wv = tid >> 5;
    int wm = wv >> 1, wn = wv & 1;          // 4 waves along M, 2 along N
    int m0 = blockIdx.y * GBM, n0 = blockIdx.x * GBN;
    int koff = (lane >> 4) * 8;             // ISA f16 A-fragment half-wave K offset
    int arow = lane & 15;

    // staging assignments (fixed per thread)
    int r0 = tid >> 2, seg = tid & 3;       // A rows r0 and r0+64, W row r0; 8-half seg
    const _Float16* Ap0 = A + (size_t)(m0 + r0) * K + seg * 8;
    const _Float16* Ap1 = A + (size_t)(m0 + r0 + 64) * K + seg * 8;
    const _Float16* Wp  = W + (size_t)(n0 + r0) * K + seg * 8;

    v8f acc[2][2] = {};

    // prologue: tile 0 into registers
    v8h ra0 = *(const v8h*)Ap0;
    v8h ra1 = *(const v8h*)Ap1;
    v8h rwc = *(const v8h*)Wp;

    for (int k0 = 0; k0 < K; k0 += GBK) {
        __syncthreads();                    // previous tile's LDS reads done
        *(v8h*)&As[r0 * GBK + seg * 8] = ra0;
        *(v8h*)&As[(r0 + 64) * GBK + seg * 8] = ra1;
        #pragma unroll
        for (int j = 0; j < 8; ++j) Bt[(seg * 8 + j) * GBTS + r0] = rwc[j];
        __syncthreads();

        int kn = k0 + GBK;
        if (kn < K) {                       // prefetch next tile into registers
            ra0 = *(const v8h*)&Ap0[kn];
            ra1 = *(const v8h*)&Ap1[kn];
            rwc = *(const v8h*)&Wp[kn];
            __builtin_prefetch(&Wp[kn + GBK], 0, 3); // WGP-scope hint
        }

        #pragma unroll
        for (int mi = 0; mi < 2; ++mi) {
            U16 a; int rb = (wm * 32 + mi * 16 + arow) * GBK;
            a.h[0] = *(const v8h*)&As[rb + koff];
            a.h[1] = *(const v8h*)&As[rb + koff + 16];
            #pragma unroll
            for (int ni = 0; ni < 2; ++ni) {
                U16 b; int nb = wn * 32 + ni * 16 + arow; // lane%16 = N column
                #pragma unroll
                for (int j = 0; j < 8; ++j) {
                    b.v[j] = Bt[(koff + j) * GBTS + nb];
                    b.v[8 + j] = Bt[(koff + 16 + j) * GBTS + nb];
                }
                acc[mi][ni] = __builtin_amdgcn_wmma_f32_16x16x32_f16(
                    false, a.v, false, b.v, (short)0, acc[mi][ni], false, false);
            }
        }
    }

    // Epilogue: C fragment layout — lane<16: (M=r, N=lane); lane>=16: (M=8+r, N=lane-16)
    int chalf = (lane >> 4) * 8, ccol = lane & 15;
    #pragma unroll
    for (int mi = 0; mi < 2; ++mi)
        #pragma unroll
        for (int ni = 0; ni < 2; ++ni)
            #pragma unroll
            for (int r = 0; r < 8; ++r) {
                int gr = m0 + wm * 32 + mi * 16 + chalf + r;
                int gc = n0 + wn * 32 + ni * 16 + ccol;
                float v = acc[mi][ni][r];
                if (bias) v += bias[gc];
                if (gelu) v = gelu_exact(v);
                size_t o = (size_t)gr * N + gc;
                if (res) v += res[o];
                if (out32) out32[o] = v;
                if (out16) out16[o] = (_Float16)v;
            }
}

// ---------------------------- Flash attention ------------------------------
// grid = (N/128, B*H). Each wave owns 16 query rows x full 64 head dims.
// Next key-block (K^T rows + V rows) is prefetched into registers while the
// current block's S/softmax/PV pipeline runs.
#define FQB 128
#define FKTS 72
__global__ __launch_bounds__(256) void attn_kernel(
    const _Float16* __restrict__ q, const _Float16* __restrict__ k,
    const _Float16* __restrict__ v, const float* __restrict__ temp,
    const float* __restrict__ hw, float* __restrict__ o) {
    __shared__ _Float16 Qs[FQB * 64];       // 16 KB
    __shared__ _Float16 Kt[64 * FKTS];      // 9 KB  transposed: Kt[d][key]
    __shared__ _Float16 Vs[64 * FKTS];      // 9 KB  row-major keys x dims
    __shared__ _Float16 Ps[8 * 16 * 64];    // 16 KB per-wave P scratch

    int tid = threadIdx.x, lane = tid & 31, wv = tid >> 5;
    int bh = blockIdx.y, bb = bh / 12, h = bh % 12;
    int q0 = blockIdx.x * FQB;
    size_t seqbase = (size_t)bb * 1024;

    // Stage Q tile once: 128 rows x 64 halves
    #pragma unroll
    for (int it = 0; it < 4; ++it) {
        int idx = tid + it * 256;           // 0..1023
        int r = idx >> 3, seg = idx & 7;
        *(v8h*)&Qs[r * 64 + seg * 8] =
            *(const v8h*)&q[(seqbase + q0 + r) * 768 + h * 64 + seg * 8];
    }

    int koff = (lane >> 4) * 8;
    int arow = lane & 15;
    int chalf = (lane >> 4) * 8, ccol = lane & 15;

    // K/V staging assignment (fixed per thread): keys fk and fk+32, 8-half seg
    int fk = tid >> 3, fseg = tid & 7;
    const _Float16* kp = k + (seqbase + fk) * 768 + h * 64 + fseg * 8;
    const _Float16* vp = v + (seqbase + fk) * 768 + h * 64 + fseg * 8;
    const size_t KSTRIDE = (size_t)32 * 768;    // 32 keys between the two chunks

    // prologue: key block 0 into registers
    v8h rk0 = *(const v8h*)kp;
    v8h rk1 = *(const v8h*)&kp[KSTRIDE];
    v8h rv0 = *(const v8h*)vp;
    v8h rv1 = *(const v8h*)&vp[KSTRIDE];

    __syncthreads();                        // Qs visible
    U16 aq[2];
    #pragma unroll
    for (int kt = 0; kt < 2; ++kt) {
        int base = (wv * 16 + arow) * 64 + kt * 32;
        aq[kt].h[0] = *(const v8h*)&Qs[base + koff];
        aq[kt].h[1] = *(const v8h*)&Qs[base + koff + 16];
    }

    float qscale = 0.125f * temp[h];        // 1/sqrt(64) * temp
    v8f of[4] = {};
    float m_[8], l_[8];
    #pragma unroll
    for (int r = 0; r < 8; ++r) { m_[r] = -1e30f; l_[r] = 0.f; }

    for (int kb = 0; kb < 16; ++kb) {
        __syncthreads();                    // previous block's LDS reads done
        #pragma unroll
        for (int j = 0; j < 8; ++j) {
            Kt[(fseg * 8 + j) * FKTS + fk]      = rk0[j];
            Kt[(fseg * 8 + j) * FKTS + fk + 32] = rk1[j];
        }
        *(v8h*)&Vs[fk * FKTS + fseg * 8]        = rv0;
        *(v8h*)&Vs[(fk + 32) * FKTS + fseg * 8] = rv1;
        __syncthreads();

        if (kb + 1 < 16) {                  // prefetch next key block into regs
            size_t nb = (size_t)(kb + 1) * 64 * 768;
            rk0 = *(const v8h*)&kp[nb];
            rk1 = *(const v8h*)&kp[nb + KSTRIDE];
            rv0 = *(const v8h*)&vp[nb];
            rv1 = *(const v8h*)&vp[nb + KSTRIDE];
        }

        // S = (Q*scale) K^T : 16x64 per wave, 4 n-tiles
        v8f s[4];
        #pragma unroll
        for (int nt = 0; nt < 4; ++nt) {
            v8f sc = {};
            #pragma unroll
            for (int kt = 0; kt < 2; ++kt) {
                U16 b; int base = (kt * 32 + koff) * FKTS + nt * 16 + ccol;
                #pragma unroll
                for (int j = 0; j < 8; ++j) {
                    b.v[j] = Kt[base + j * FKTS];
                    b.v[8 + j] = Kt[base + (16 + j) * FKTS];
                }
                sc = __builtin_amdgcn_wmma_f32_16x16x32_f16(
                    false, aq[kt].v, false, b.v, (short)0, sc, false, false);
            }
            #pragma unroll
            for (int r = 0; r < 8; ++r) sc[r] *= qscale;
            s[nt] = sc;
        }

        // Online softmax per row (each row lives in one 16-lane half)
        #pragma unroll
        for (int r = 0; r < 8; ++r) {
            float mn = fmaxf(fmaxf(s[0][r], s[1][r]), fmaxf(s[2][r], s[3][r]));
            mn = fmaxf(mn, __shfl_xor(mn, 1));
            mn = fmaxf(mn, __shfl_xor(mn, 2));
            mn = fmaxf(mn, __shfl_xor(mn, 4));
            mn = fmaxf(mn, __shfl_xor(mn, 8));
            float mnew = fmaxf(m_[r], mn);
            float alpha = __expf(m_[r] - mnew);
            m_[r] = mnew; l_[r] *= alpha;
            of[0][r] *= alpha; of[1][r] *= alpha; of[2][r] *= alpha; of[3][r] *= alpha;
            float ps = 0.f;
            #pragma unroll
            for (int nt = 0; nt < 4; ++nt) {
                float p = __expf(s[nt][r] - mnew);
                s[nt][r] = p; ps += p;
            }
            ps += __shfl_xor(ps, 1); ps += __shfl_xor(ps, 2);
            ps += __shfl_xor(ps, 4); ps += __shfl_xor(ps, 8);
            l_[r] += ps;
        }

        // C-layout -> A-layout via per-wave LDS scratch (same-wave DS ordering)
        #pragma unroll
        for (int nt = 0; nt < 4; ++nt)
            #pragma unroll
            for (int r = 0; r < 8; ++r)
                Ps[wv * 1024 + (chalf + r) * 64 + nt * 16 + ccol] = (_Float16)s[nt][r];

        // O += P V
        #pragma unroll
        for (int kt = 0; kt < 2; ++kt) {
            U16 ap; int base = wv * 1024 + arow * 64 + kt * 32;
            ap.h[0] = *(const v8h*)&Ps[base + koff];
            ap.h[1] = *(const v8h*)&Ps[base + koff + 16];
            #pragma unroll
            for (int nt = 0; nt < 4; ++nt) {
                U16 bv; int vb = (kt * 32 + koff) * FKTS + nt * 16 + ccol;
                #pragma unroll
                for (int j = 0; j < 8; ++j) {
                    bv.v[j] = Vs[vb + j * FKTS];
                    bv.v[8 + j] = Vs[vb + (16 + j) * FKTS];
                }
                of[nt] = __builtin_amdgcn_wmma_f32_16x16x32_f16(
                    false, ap.v, false, bv.v, (short)0, of[nt], false, false);
            }
        }
    }

    // softmax(hw) over 12 heads, folded into final scale (post-softmax constant)
    float hmx = hw[0];
    for (int i = 1; i < 12; ++i) hmx = fmaxf(hmx, hw[i]);
    float hden = 0.f;
    for (int i = 0; i < 12; ++i) hden += __expf(hw[i] - hmx);
    float hsc = __expf(hw[h] - hmx) / hden;

    #pragma unroll
    for (int r = 0; r < 8; ++r) {
        float inv = hsc / l_[r];
        size_t grow = (seqbase + q0 + wv * 16 + chalf + r) * 768 + h * 64;
        #pragma unroll
        for (int nt = 0; nt < 4; ++nt)
            o[grow + nt * 16 + ccol] = of[nt][r] * inv;
    }
}

// -------------------- multi-scale grouped conv branch ----------------------
// acc[t,c] = o[t,c] + (y1+y3+y5)[t,c]/3 ; groups=H=12, group width 64.
__global__ __launch_bounds__(256) void conv_kernel(
    const float* __restrict__ o,
    const float* __restrict__ w1, const float* __restrict__ w3, const float* __restrict__ w5,
    const float* __restrict__ b1, const float* __restrict__ b3, const float* __restrict__ b5,
    _Float16* __restrict__ acc) {
    int idx = blockIdx.x * 256 + threadIdx.x;
    if (idx >= 4096 * 768) return;
    int c = idx % 768, t = idx / 768;
    int n = t & 1023, bb = t >> 10;
    int g = c >> 6;
    const float* ob = o + ((size_t)bb * 1024) * 768 + g * 64;
    float ysum = b1[c] + b3[c] + b5[c];
    { // s = 1
        const float* w = w1 + (size_t)c * 64;
        const float* xr = ob + (size_t)n * 768;
        #pragma unroll 8
        for (int i = 0; i < 64; ++i) ysum += w[i] * xr[i];
    }
    for (int tap = 0; tap < 3; ++tap) { // s = 3, pad 1
        int nn = n + tap - 1;
        if (nn < 0 || nn >= 1024) continue;
        const float* xr = ob + (size_t)nn * 768;
        const float* w = w3 + (size_t)c * 192 + tap;
        #pragma unroll 8
        for (int i = 0; i < 64; ++i) ysum += w[i * 3] * xr[i];
    }
    for (int tap = 0; tap < 5; ++tap) { // s = 5, pad 2
        int nn = n + tap - 2;
        if (nn < 0 || nn >= 1024) continue;
        const float* xr = ob + (size_t)nn * 768;
        const float* w = w5 + (size_t)c * 320 + tap;
        #pragma unroll 8
        for (int i = 0; i < 64; ++i) ysum += w[i * 5] * xr[i];
    }
    float val = o[(size_t)t * 768 + c] + ysum * (1.0f / 3.0f);
    acc[(size_t)t * 768 + c] = (_Float16)val;
}

// ===========================================================================
extern "C" void kernel_launch(void* const* d_in, const int* in_sizes, int n_in,
                              void* d_out, int out_size, void* d_ws, size_t ws_size,
                              hipStream_t stream) {
    const int T = 4096, D = 768, VD = 1024, MLP = 3072, AD = 512, DEPTH = 4;

    // --- input mapping: JAX sorted-dict tree_leaves order for params; ------
    // top-level may be video-first (insertion order) or params-first (sorted).
    int P; const float* video;
    if (in_sizes[0] == 4 * 1024 * 1024) { video = (const float*)d_in[0]; P = 1; }
    else                                { video = (const float*)d_in[n_in - 1]; P = 0; }
    auto IN = [&](int i) -> const float* { return (const float*)d_in[i]; };
    const float* dec1_b = IN(P + 0); const float* dec1_w = IN(P + 1);
    const float* dec2_b = IN(P + 2); const float* dec2_w = IN(P + 3);
    const float* dln_b  = IN(P + 4); const float* dln_g  = IN(P + 5);
    const float* emb_b  = IN(P + 6); const float* emb_w  = IN(P + 7);
    const float* eln_b  = IN(P + 8); const float* eln_g  = IN(P + 9);
    // per-layer leaf order (sorted): conv_b[3], conv_w[3], hw, out.b, out.w,
    //                                temp, wk, wq, wv, ff1.b, ff1.w, ff2.b,
    //                                ff2.w, ln1.b, ln1.g, ln2.b, ln2.g
    auto LIX = [&](int li) { return P + 10 + li * 21; };

    // --- workspace bump allocator ------------------------------------------
    size_t off = 0;
    auto alloc = [&](size_t bytes) -> void* {
        void* p = (char*)d_ws + off;
        off = (off + bytes + 255) & ~(size_t)255;
        return p;
    };
    _Float16* we16   = (_Float16*)alloc((size_t)D * VD * 2);
    _Float16* wq16[4], *wk16[4], *wv16[4], *wo16[4], *f1w16[4], *f2w16[4];
    for (int li = 0; li < DEPTH; ++li) {
        wq16[li]  = (_Float16*)alloc((size_t)D * D * 2);
        wk16[li]  = (_Float16*)alloc((size_t)D * D * 2);
        wv16[li]  = (_Float16*)alloc((size_t)D * D * 2);
        wo16[li]  = (_Float16*)alloc((size_t)D * D * 2);
        f1w16[li] = (_Float16*)alloc((size_t)MLP * D * 2);
        f2w16[li] = (_Float16*)alloc((size_t)D * MLP * 2);
    }
    _Float16* d1w16 = (_Float16*)alloc((size_t)2 * D * D * 2);
    _Float16* d2w16 = (_Float16*)alloc((size_t)AD * 2 * D * 2);
    _Float16* vf16  = (_Float16*)alloc((size_t)T * VD * 2);
    float*    x32   = (float*)   alloc((size_t)T * D * 4);     // residual stream
    float*    scr32 = (float*)   alloc((size_t)T * 2 * D * 4); // y/o/dec1 scratch
    _Float16* h16   = (_Float16*)alloc((size_t)T * D * 2);     // LN out / acc16 / x16
    _Float16* q16   = (_Float16*)alloc((size_t)T * D * 2);
    _Float16* k16   = (_Float16*)alloc((size_t)T * D * 2);
    _Float16* v16   = (_Float16*)alloc((size_t)T * D * 2);
    _Float16* g16   = (_Float16*)alloc((size_t)T * MLP * 2);   // ff1 out / dh16
    (void)ws_size;

    auto castW = [&](const float* src, _Float16* dst, int n) {
        cast_f16_kernel<<<dim3((n + 255) / 256), dim3(256), 0, stream>>>(src, dst, n);
    };
    auto gemm = [&](const _Float16* A, const _Float16* W, const float* bias,
                    const float* res, float* o32, _Float16* o16,
                    int M, int N, int K, int gelu) {
        gemm_nt_kernel<<<dim3(N / 64, M / 128), dim3(256), 0, stream>>>(
            A, W, bias, res, o32, o16, M, N, K, gelu);
    };
    auto ln = [&](const float* x, int C, const float* g, const float* b,
                  float* o32, _Float16* o16, int addPE, int gelu) {
        ln_kernel<<<dim3(T), dim3(256), 0, stream>>>(x, C, g, b, o32, o16, addPE, gelu);
    };

    // --- weight casts (every call; deterministic) --------------------------
    castW(emb_w, we16, D * VD);
    for (int li = 0; li < DEPTH; ++li) {
        int L = LIX(li);
        castW(IN(L + 11), wq16[li], D * D);   // wq
        castW(IN(L + 10), wk16[li], D * D);   // wk
        castW(IN(L + 12), wv16[li], D * D);   // wv
        castW(IN(L + 8),  wo16[li], D * D);   // out.w
        castW(IN(L + 14), f1w16[li], MLP * D);
        castW(IN(L + 16), f2w16[li], D * MLP);
    }
    castW(dec1_w, d1w16, 2 * D * D);
    castW(dec2_w, d2w16, AD * 2 * D);

    // --- encode_video: L2 norm -> embed -> LN -> +PE -----------------------
    l2norm_kernel<<<dim3(T), dim3(256), 0, stream>>>(video, vf16, VD);
    gemm(vf16, we16, emb_b, nullptr, scr32, nullptr, T, D, VD, 0);
    ln(scr32, D, eln_g, eln_b, x32, nullptr, /*PE*/1, 0);

    // --- transformer layers ------------------------------------------------
    for (int li = 0; li < DEPTH; ++li) {
        int L = LIX(li);
        ln(x32, D, IN(L + 18), IN(L + 17), nullptr, h16, 0, 0);       // ln1
        gemm(h16, wq16[li], nullptr, nullptr, nullptr, q16, T, D, D, 0);
        gemm(h16, wk16[li], nullptr, nullptr, nullptr, k16, T, D, D, 0);
        gemm(h16, wv16[li], nullptr, nullptr, nullptr, v16, T, D, D, 0);
        attn_kernel<<<dim3(1024 / FQB, 4 * 12), dim3(256), 0, stream>>>(
            q16, k16, v16, IN(L + 9) /*temp*/, IN(L + 6) /*hw*/, scr32);
        conv_kernel<<<dim3((T * D + 255) / 256), dim3(256), 0, stream>>>(
            scr32, IN(L + 3), IN(L + 4), IN(L + 5),               // conv_w s=1,3,5
            IN(L + 0), IN(L + 1), IN(L + 2), h16);                // conv_b, acc16
        gemm(h16, wo16[li], IN(L + 7), x32, x32, nullptr, T, D, D, 0); // out proj + res
        ln(x32, D, IN(L + 20), IN(L + 19), nullptr, h16, 0, 0);   // ln2
        gemm(h16, f1w16[li], IN(L + 13), nullptr, nullptr, g16, T, MLP, D, /*gelu*/1);
        gemm(g16, f2w16[li], IN(L + 15), x32, x32, nullptr, T, D, MLP, 0);
    }

    // --- audio decoder -----------------------------------------------------
    cast32to16_kernel<<<dim3((T * D + 255) / 256), dim3(256), 0, stream>>>(x32, h16, T * D);
    gemm(h16, d1w16, dec1_b, nullptr, scr32, nullptr, T, 2 * D, D, 0);
    ln(scr32, 2 * D, dln_g, dln_b, nullptr, g16 /*dh16*/, 0, /*gelu*/1);
    gemm(g16, d2w16, dec2_b, nullptr, (float*)d_out, nullptr, T, AD, 2 * D, 0);
    (void)out_size;
}
